// GNNEncoder_15118284882769
// MI455X (gfx1250) — matrix operand order repaired and checked by reading, not compiled, not run on hardware
//
#include <hip/hip_runtime.h>
#include <hip/hip_bf16.h>

// ---------------------------------------------------------------------------
// Types for WMMA fragments (CDNA5 / gfx1250, wave32)
// ---------------------------------------------------------------------------
typedef __attribute__((ext_vector_type(16))) __bf16        bf16x16;
typedef __attribute__((ext_vector_type(8)))  float         f32x8;
typedef __attribute__((ext_vector_type(4)))  float         f32x4;
typedef __attribute__((ext_vector_type(4)))  unsigned int  u32x4;

union FragAB {
    bf16x16        bf;
    unsigned short us[16];
    u32x4          q[2];
};

__device__ __forceinline__ unsigned short f2bf(float f) {
    unsigned int u = __float_as_uint(f);
    u += 0x7FFFu + ((u >> 16) & 1u);   // round-to-nearest-even
    return (unsigned short)(u >> 16);
}

#define WMMA_BF16(A, B, C) \
    __builtin_amdgcn_wmma_f32_16x16x32_bf16(false, (A), false, (B), (short)0, (C), false, false)

// Problem constants
static constexpr int NODES = 128;
static constexpr int NFEAT = 256;
static constexpr int BATCH = 512;
static constexpr int KFC   = NODES * NFEAT;   // 32768
static constexpr int KSPLIT = 8;              // K-chunks of 4096 in fc GEMM

// ---------------------------------------------------------------------------
// Prep: gcn_w (F,G) f32  ->  wT (G,F) bf16   (transpose + convert)
// ---------------------------------------------------------------------------
__global__ __launch_bounds__(256) void prep_gcnw(const float* __restrict__ w,
                                                 unsigned short* __restrict__ wT) {
    int f = blockIdx.x;      // 0..255
    int g = threadIdx.x;     // 0..255
    wT[g * NFEAT + f] = f2bf(w[f * NFEAT + g]);
}

// ---------------------------------------------------------------------------
// Prep: fc_w f32 -> bf16 (same layout, (256, 32768) row-major, K contiguous)
// ---------------------------------------------------------------------------
__global__ __launch_bounds__(256) void prep_fcw(const float* __restrict__ w,
                                                unsigned short* __restrict__ o) {
    size_t i4 = ((size_t)blockIdx.x * 256 + threadIdx.x) * 4;
    f32x4 v = *reinterpret_cast<const f32x4*>(w + i4);
    union { unsigned short us[4]; unsigned long long u; } st;
    st.us[0] = f2bf(v.x); st.us[1] = f2bf(v.y);
    st.us[2] = f2bf(v.z); st.us[3] = f2bf(v.w);
    *reinterpret_cast<unsigned long long*>(o + i4) = st.u;
}

// ---------------------------------------------------------------------------
// K1: support = x @ gcn_w      (per batch: 128x256 = (128x256)*(256x256))
//     output stored TRANSPOSED: supT[b][g][m]  (bf16)
// grid.x = 512 (batch), 256 threads = 8 waves, wave -> 16-node M-tile
// ---------------------------------------------------------------------------
__global__ __launch_bounds__(256) void gcn_support_kernel(
        const float* __restrict__ x,            // (512,128,256) f32
        const unsigned short* __restrict__ wT,  // (256 g, 256 f) bf16
        unsigned short* __restrict__ supT) {    // (512, 256 g, 128 m) bf16
    const int b    = blockIdx.x;
    const int lane = threadIdx.x & 31;
    const int wave = threadIdx.x >> 5;      // M-tile index 0..7
    const int lm   = lane & 15;
    const int hi   = lane >> 4;

    // Preload A fragments for all 8 K-steps (row = node, K contiguous in x)
    const float* xrow = x + ((size_t)b * NODES + wave * 16 + lm) * NFEAT;
    FragAB afrag[8];
#pragma unroll
    for (int kk = 0; kk < 8; ++kk) {
        const int kb = kk * 32 + hi * 8;
        f32x4 p0 = *reinterpret_cast<const f32x4*>(xrow + kb);
        f32x4 p1 = *reinterpret_cast<const f32x4*>(xrow + kb + 4);
        f32x4 p2 = *reinterpret_cast<const f32x4*>(xrow + kb + 16);
        f32x4 p3 = *reinterpret_cast<const f32x4*>(xrow + kb + 20);
        afrag[kk].us[0]  = f2bf(p0.x); afrag[kk].us[1]  = f2bf(p0.y);
        afrag[kk].us[2]  = f2bf(p0.z); afrag[kk].us[3]  = f2bf(p0.w);
        afrag[kk].us[4]  = f2bf(p1.x); afrag[kk].us[5]  = f2bf(p1.y);
        afrag[kk].us[6]  = f2bf(p1.z); afrag[kk].us[7]  = f2bf(p1.w);
        afrag[kk].us[8]  = f2bf(p2.x); afrag[kk].us[9]  = f2bf(p2.y);
        afrag[kk].us[10] = f2bf(p2.z); afrag[kk].us[11] = f2bf(p2.w);
        afrag[kk].us[12] = f2bf(p3.x); afrag[kk].us[13] = f2bf(p3.y);
        afrag[kk].us[14] = f2bf(p3.z); afrag[kk].us[15] = f2bf(p3.w);
    }

#pragma unroll 1
    for (int nt = 0; nt < 16; ++nt) {
        f32x8 acc = {};
        // B fragment: n = nt*16+lm fixed per lane, K contiguous in wT row
        const unsigned short* wrow = wT + (nt * 16 + lm) * NFEAT + hi * 8;
#pragma unroll
        for (int kk = 0; kk < 8; ++kk) {
            FragAB bfrag;
            const u32x4* wq = reinterpret_cast<const u32x4*>(wrow + kk * 32);
            bfrag.q[0] = wq[0];
            bfrag.q[1] = wq[2];
            acc = WMMA_BF16(afrag[kk].bf, bfrag.bf, acc);
        }
        // Transposed store: supT[b][g = nt*16+lm][m = wave*16 + hi*8 + r], r contiguous
        union { unsigned short us[8]; u32x4 q; } st;
#pragma unroll
        for (int r = 0; r < 8; ++r) st.us[r] = f2bf(acc[r]);
        unsigned short* dst = supT +
            (((size_t)b * NFEAT + nt * 16 + lm) * NODES + wave * 16 + hi * 8);
        *reinterpret_cast<u32x4*>(dst) = st.q;
    }
}

// ---------------------------------------------------------------------------
// K2: mid = adj @ support + gcn_b   (per batch: (128x128)*(128x256))
//     mid stored row-major (b, node, g) bf16  == the "flat" layout for K3
// ---------------------------------------------------------------------------
__global__ __launch_bounds__(256) void gcn_aggregate_kernel(
        const float* __restrict__ adj,           // (512,128,128) f32
        const unsigned short* __restrict__ supT, // (512,256,128) bf16
        const float* __restrict__ gb,            // (256,)
        unsigned short* __restrict__ mid) {      // (512,128,256) bf16
    const int b    = blockIdx.x;
    const int lane = threadIdx.x & 31;
    const int wave = threadIdx.x >> 5;
    const int lm   = lane & 15;
    const int hi   = lane >> 4;

    const float* arow = adj + ((size_t)b * NODES + wave * 16 + lm) * NODES;
    FragAB afrag[4];
#pragma unroll
    for (int kk = 0; kk < 4; ++kk) {
        const int kb = kk * 32 + hi * 8;
        f32x4 p0 = *reinterpret_cast<const f32x4*>(arow + kb);
        f32x4 p1 = *reinterpret_cast<const f32x4*>(arow + kb + 4);
        f32x4 p2 = *reinterpret_cast<const f32x4*>(arow + kb + 16);
        f32x4 p3 = *reinterpret_cast<const f32x4*>(arow + kb + 20);
        afrag[kk].us[0]  = f2bf(p0.x); afrag[kk].us[1]  = f2bf(p0.y);
        afrag[kk].us[2]  = f2bf(p0.z); afrag[kk].us[3]  = f2bf(p0.w);
        afrag[kk].us[4]  = f2bf(p1.x); afrag[kk].us[5]  = f2bf(p1.y);
        afrag[kk].us[6]  = f2bf(p1.z); afrag[kk].us[7]  = f2bf(p1.w);
        afrag[kk].us[8]  = f2bf(p2.x); afrag[kk].us[9]  = f2bf(p2.y);
        afrag[kk].us[10] = f2bf(p2.z); afrag[kk].us[11] = f2bf(p2.w);
        afrag[kk].us[12] = f2bf(p3.x); afrag[kk].us[13] = f2bf(p3.y);
        afrag[kk].us[14] = f2bf(p3.z); afrag[kk].us[15] = f2bf(p3.w);
    }

#pragma unroll 1
    for (int nt = 0; nt < 16; ++nt) {
        f32x8 acc = {};
        const unsigned short* srow = supT +
            ((size_t)b * NFEAT + nt * 16 + lm) * NODES + hi * 8;
#pragma unroll
        for (int kk = 0; kk < 4; ++kk) {
            FragAB bfrag;
            const u32x4* sq = reinterpret_cast<const u32x4*>(srow + kk * 32);
            bfrag.q[0] = sq[0];
            bfrag.q[1] = sq[2];
            acc = WMMA_BF16(afrag[kk].bf, bfrag.bf, acc);
        }
        const float bias = gb[nt * 16 + lm];     // per-column (g) bias
        // mid[b][node = wave*16 + hi*8 + r][g = nt*16 + lm]
        unsigned short* base = mid +
            ((size_t)b * NODES + wave * 16 + hi * 8) * NFEAT + nt * 16 + lm;
#pragma unroll
        for (int r = 0; r < 8; ++r) base[(size_t)r * NFEAT] = f2bf(acc[r] + bias);
    }
}

// ---------------------------------------------------------------------------
// K3: fc partials with 2x2 register blocking.
// out = flat @ fc_w.T ; M=512, N=256, K=32768 split 8 ways (chunks of 4096).
// Wave tile = 32x32 (4 accumulators). Tasks = 8 ks * 16 mt2 * 8 nt2 = 1024
// waves -> 256 blocks * 128 threads (4 waves/block).
// Per inner iter: 8 x b128 loads feed 4 WMMAs (2x better bytes/FLOP than 1x1).
// ---------------------------------------------------------------------------
__global__ __launch_bounds__(128) void fc_partial_kernel(
        const unsigned short* __restrict__ flat,  // (512, 32768) bf16
        const unsigned short* __restrict__ fw,    // (256, 32768) bf16
        float* __restrict__ part) {               // (8, 512, 256) f32
    const int task = blockIdx.x * 4 + (threadIdx.x >> 5);  // 0..1023
    const int ks   = task >> 7;          // 0..7
    const int rem  = task & 127;
    const int mt2  = rem >> 3;           // 0..15  (32-row block)
    const int nt2  = rem & 7;            // 0..7   (32-col block)
    const int lane = threadIdx.x & 31;
    const int lm   = lane & 15;
    const int hi   = lane >> 4;

    const unsigned short* arow0 = flat + ((size_t)(mt2 * 32 + lm)) * KFC + (size_t)ks * 4096;
    const unsigned short* arow1 = arow0 + (size_t)16 * KFC;
    const unsigned short* brow0 = fw + ((size_t)(nt2 * 32 + lm)) * KFC + (size_t)ks * 4096 + hi * 8;
    const unsigned short* brow1 = brow0 + (size_t)16 * KFC;

    f32x8 acc00 = {}, acc01 = {}, acc10 = {}, acc11 = {};
#pragma unroll 2
    for (int kk = 0; kk < 128; ++kk) {       // 128 * K32 = 4096
        const int kb = kk * 32;
        // pull the next lines toward L0 while current WMMAs run
        __builtin_prefetch((const void*)(arow0 + kb + 2048), 0, 0);
        __builtin_prefetch((const void*)(brow0 + kb + 2048), 0, 0);
        FragAB a0, a1, b0, b1;
        a0.q[0] = *reinterpret_cast<const u32x4*>(arow0 + kb + hi * 8);
        a0.q[1] = *reinterpret_cast<const u32x4*>(arow0 + kb + 16 + hi * 8);
        a1.q[0] = *reinterpret_cast<const u32x4*>(arow1 + kb + hi * 8);
        a1.q[1] = *reinterpret_cast<const u32x4*>(arow1 + kb + 16 + hi * 8);
        const u32x4* bq0 = reinterpret_cast<const u32x4*>(brow0 + kb);
        const u32x4* bq1 = reinterpret_cast<const u32x4*>(brow1 + kb);
        b0.q[0] = bq0[0]; b0.q[1] = bq0[2];
        b1.q[0] = bq1[0]; b1.q[1] = bq1[2];
        acc00 = WMMA_BF16(a0.bf, b0.bf, acc00);
        acc01 = WMMA_BF16(a0.bf, b1.bf, acc01);
        acc10 = WMMA_BF16(a1.bf, b0.bf, acc10);
        acc11 = WMMA_BF16(a1.bf, b1.bf, acc11);
    }
    // part[ks][row][col]; fragment rows = base + hi*8 + r, cols = base + lm
    float* dst = part + ((size_t)ks * BATCH + mt2 * 32 + hi * 8) * NFEAT + nt2 * 32 + lm;
#pragma unroll
    for (int r = 0; r < 8; ++r) {
        dst[(size_t)r * NFEAT]            = acc00[r];
        dst[(size_t)r * NFEAT + 16]       = acc01[r];
        dst[(size_t)(r + 16) * NFEAT]     = acc10[r];
        dst[(size_t)(r + 16) * NFEAT + 16]= acc11[r];
    }
}

// ---------------------------------------------------------------------------
// K3b: deterministic tree-reduce of 8 partials + fc_b -> d_out segments 0, 2
// ---------------------------------------------------------------------------
__global__ __launch_bounds__(256) void fc_reduce_kernel(
        const float* __restrict__ part, const float* __restrict__ fb,
        float* __restrict__ out) {
    const int idx = blockIdx.x * 256 + threadIdx.x;   // 0..131071
    const int SEG = BATCH * NFEAT;                    // 131072
    float s0 = part[idx]           + part[SEG + idx];
    float s1 = part[2 * SEG + idx] + part[3 * SEG + idx];
    float s2 = part[4 * SEG + idx] + part[5 * SEG + idx];
    float s3 = part[6 * SEG + idx] + part[7 * SEG + idx];
    float s  = fb[idx & (NFEAT - 1)] + ((s0 + s1) + (s2 + s3));
    out[idx]           = s;   // output (segment 0)
    out[2 * SEG + idx] = s;   // output again (segment 2)
}

// ---------------------------------------------------------------------------
// K4: useful_input = x[:,0,:] -> d_out segment 1
// ---------------------------------------------------------------------------
__global__ __launch_bounds__(256) void copy_useful_kernel(
        const float* __restrict__ x, float* __restrict__ out) {
    const int idx = blockIdx.x * 256 + threadIdx.x;   // 0..131071
    const int b = idx >> 8;
    const int f = idx & (NFEAT - 1);
    out[BATCH * NFEAT + idx] = x[(size_t)b * NODES * NFEAT + f];
}

// ---------------------------------------------------------------------------
// Launch
// ---------------------------------------------------------------------------
extern "C" void kernel_launch(void* const* d_in, const int* in_sizes, int n_in,
                              void* d_out, int out_size, void* d_ws, size_t ws_size,
                              hipStream_t stream) {
    const float* x     = (const float*)d_in[0];   // (512,128,256)
    const float* adj   = (const float*)d_in[1];   // (512,128,128)
    const float* gcn_w = (const float*)d_in[2];   // (256,256)
    const float* gcn_b = (const float*)d_in[3];   // (256,)
    const float* fc_w  = (const float*)d_in[4];   // (256,32768)
    const float* fc_b  = (const float*)d_in[5];   // (256,)
    float* out = (float*)d_out;

    // Workspace layout (bytes, 256B-aligned regions)
    char* ws = (char*)d_ws;
    unsigned short* wT    = (unsigned short*)(ws);                       //   128 KiB
    unsigned short* fwbf  = (unsigned short*)(ws + (size_t)131072);      //    16 MiB
    unsigned short* supT  = (unsigned short*)(ws + (size_t)16908288);    //    32 MiB
    unsigned short* mid   = (unsigned short*)(ws + (size_t)50462720);    //    32 MiB
    float*          part  = (float*)        (ws + (size_t)84017152);     //     4 MiB (8 partials)

    prep_gcnw           <<<256,  256, 0, stream>>>(gcn_w, wT);
    prep_fcw            <<<8192, 256, 0, stream>>>(fc_w, fwbf);
    gcn_support_kernel  <<<512,  256, 0, stream>>>(x, wT, supT);
    gcn_aggregate_kernel<<<512,  256, 0, stream>>>(adj, supT, gcn_b, mid);
    fc_partial_kernel   <<<256,  128, 0, stream>>>(mid, fwbf, part);
    fc_reduce_kernel    <<<512,  256, 0, stream>>>(part, fc_b, out);
    copy_useful_kernel  <<<512,  256, 0, stream>>>(x, out);
}
